// MTGRU_56770877718784
// MI455X (gfx1250) — matrix operand order repaired and checked by reading, not compiled, not run on hardware
//
#include <hip/hip_runtime.h>
#include <hip/hip_bf16.h>

typedef __bf16 bf16;
typedef __attribute__((ext_vector_type(16))) __bf16 v16bf;
typedef __attribute__((ext_vector_type(8)))  __bf16 v8bf;
typedef __attribute__((ext_vector_type(8)))  float  v8f;

// ---- problem constants --------------------------------------------------
constexpr int Bc   = 32;
constexpr int Tc   = 100;
constexpr int INc  = 64;
constexpr int Kc   = 10;
constexpr int Hc   = 512;
constexpr int NPLc = 30;
constexpr int HUc  = 64;          // HUMAN
constexpr int G3   = 3 * Hc;      // 1536
constexpr int S_Whh_ = Hc * G3;   // 786432
constexpr int S_bh_  = G3;        // 1536
constexpr int S_Wih_ = INc * G3;  // 98304
constexpr int S_C_   = Hc * HUc;  // 32768
constexpr int S_D_   = INc * HUc; // 4096
constexpr int Pp = S_Whh_ + S_bh_ + S_Wih_ + S_C_ + S_D_; // 923136

// ---- WMMA helpers (layouts per CDNA5 ISA §7.12.2) ------------
__device__ __forceinline__ v8f wmma_bf16(v16bf a, v16bf b, v8f c) {
  return __builtin_amdgcn_wmma_f32_16x16x32_bf16(false, a, false, b,
                                                 (short)0, c, false, false);
}

// A (16x32 bf16) from row-major source, row stride ld. lane L: row m=L&15,
// half=L>>4; element e -> K = (e<8 ? e : e+8) + half*8.  Two b128 loads.
__device__ __forceinline__ v16bf load_A(const bf16* base, int ld, int mlim) {
  int lane = threadIdx.x & 31;
  int m = lane & 15, half = lane >> 4;
  v16bf a;
  if (m < mlim) {
    const bf16* row = base + (size_t)m * ld + half * 8;
    v8bf lo = *(const v8bf*)(row);
    v8bf hi = *(const v8bf*)(row + 16);
#pragma unroll
    for (int e = 0; e < 8; ++e) { a[e] = lo[e]; a[8 + e] = hi[e]; }
  } else {
#pragma unroll
    for (int e = 0; e < 16; ++e) a[e] = (bf16)0.0f;
  }
  return a;
}

// A with every row broadcast from the same vector (GEMV-as-GEMM trick).
__device__ __forceinline__ v16bf load_A_bcast(const bf16* vec) {
  int half = (threadIdx.x & 31) >> 4;
  const bf16* p = vec + half * 8;
  v8bf lo = *(const v8bf*)(p);
  v8bf hi = *(const v8bf*)(p + 16);
  v16bf a;
#pragma unroll
  for (int e = 0; e < 8; ++e) { a[e] = lo[e]; a[8 + e] = hi[e]; }
  return a;
}

// B (32x16 bf16) from TRANSPOSED source: B^T stored row-major [N][K] with
// row stride ldT. lane L: N=L&15, element e -> K = (e<8 ? e : e+8) + (L>>4)*8
// = two b128 loads.
__device__ __forceinline__ v16bf load_BT(const bf16* baseT, int ldT) {
  int lane = threadIdx.x & 31;
  int n = lane & 15, kh = (lane >> 4) * 8;
  const bf16* row = baseT + (size_t)n * ldT + kh;
  v8bf lo = *(const v8bf*)(row);
  v8bf hi = *(const v8bf*)(row + 16);
  v16bf b;
#pragma unroll
  for (int e = 0; e < 8; ++e) { b[e] = lo[e]; b[8 + e] = hi[e]; }
  return b;
}

__device__ __forceinline__ float sigmoidf_(float x) {
  return 1.0f / (1.0f + expf(-x));
}

// ---- K0a: plain f32 -> bf16 convert -------------------------------------
__global__ void cvt_bf16_kernel(const float* __restrict__ in,
                                bf16* __restrict__ out, int n) {
  int i = blockIdx.x * blockDim.x + threadIdx.x;
  if (i < n) out[i] = (bf16)in[i];
}

// ---- K0b: f32 -> bf16 transpose-convert: in [rows][cols] -> out [cols][rows]
__global__ void cvt_T_bf16_kernel(const float* __restrict__ in,
                                  bf16* __restrict__ out, int rows, int cols) {
  int i = blockIdx.x * blockDim.x + threadIdx.x;
  if (i < rows * cols) {
    int r = i / cols, c = i - r * cols;
    out[(size_t)c * rows + r] = (bf16)in[i];
  }
}

// ---- K1: hypernet front (z -> tanh MLP -> h2 (B x 32, padded)) ----------
__global__ void hyper_kernel(const float* __restrict__ mu,
                             const float* __restrict__ sd,
                             const float* __restrict__ eps,
                             const float* __restrict__ W1,
                             const float* __restrict__ b1,
                             const float* __restrict__ W2,
                             const float* __restrict__ b2,
                             float* __restrict__ h2out) {
  int b = blockIdx.x;               // one block per batch, 512 threads
  __shared__ float zsh[16];
  __shared__ float h1sh[512];
  int tid = threadIdx.x;
  if (tid < Kc) zsh[tid] = mu[b * Kc + tid] + eps[b * Kc + tid] * sd[b * Kc + tid];
  __syncthreads();
  float acc = b1[tid];
#pragma unroll
  for (int k = 0; k < Kc; ++k) acc += zsh[k] * W1[k * 512 + tid];
  h1sh[tid] = tanhf(acc);
  __syncthreads();
  if (tid < 32) {
    float a = 0.0f;
    if (tid < NPLc) {
      a = b2[tid];
      for (int j = 0; j < 512; ++j) a += h1sh[j] * W2[j * NPLc + tid];
    }
    h2out[b * 32 + tid] = a;        // pad cols 30,31 with 0
  }
}

// ---- K2: psi = h2 @ W3 + b3, scattered into TRANSPOSED typed buffers ----
__global__ void psi_kernel(const float* __restrict__ h2,
                           const float* __restrict__ W3,
                           const float* __restrict__ b3,
                           bf16*  __restrict__ whhT,  // (B, 3H, H)  = Whh^T
                           float* __restrict__ bhp,   // (B, 3H)
                           bf16*  __restrict__ wihT,  // (B, 3H, IN) = Wih^T
                           bf16*  __restrict__ CT,    // (B, 64, H)  = C^T
                           bf16*  __restrict__ DT) {  // (B, 64, IN) = D^T
  __shared__ float h2s[32 * 32];
  for (int i = threadIdx.x; i < 32 * 32; i += blockDim.x) h2s[i] = h2[i];
  __syncthreads();
  int p = blockIdx.x * blockDim.x + threadIdx.x;
  if (p >= Pp) return;
  float bias = b3[p];
  float acc[32];
#pragma unroll
  for (int b = 0; b < 32; ++b) acc[b] = bias;
  for (int k = 0; k < NPLc; ++k) {
    float w = W3[(size_t)k * Pp + p];
#pragma unroll
    for (int b = 0; b < 32; ++b) acc[b] += h2s[b * 32 + k] * w;
  }
  if (p < S_Whh_) {                         // Whh: p = h*G3 + g
    int h = p / G3, g = p - h * G3;
    size_t q = (size_t)g * Hc + h;
#pragma unroll
    for (int b = 0; b < 32; ++b) whhT[(size_t)b * S_Whh_ + q] = (bf16)acc[b];
  } else if (p < S_Whh_ + S_bh_) {
    int q = p - S_Whh_;
#pragma unroll
    for (int b = 0; b < 32; ++b) bhp[b * G3 + q] = acc[b];
  } else if (p < S_Whh_ + S_bh_ + S_Wih_) { // Wih: q = i*G3 + g
    int q0 = p - (S_Whh_ + S_bh_);
    int i = q0 / G3, g = q0 - i * G3;
    size_t q = (size_t)g * INc + i;
#pragma unroll
    for (int b = 0; b < 32; ++b) wihT[(size_t)b * S_Wih_ + q] = (bf16)acc[b];
  } else if (p < S_Whh_ + S_bh_ + S_Wih_ + S_C_) { // C: q = h*64 + o
    int q0 = p - (S_Whh_ + S_bh_ + S_Wih_);
    int h = q0 / HUc, o = q0 - h * HUc;
    size_t q = (size_t)o * Hc + h;
#pragma unroll
    for (int b = 0; b < 32; ++b) CT[(size_t)b * S_C_ + q] = (bf16)acc[b];
  } else {                                   // D: q = i*64 + o
    int q0 = p - (S_Whh_ + S_bh_ + S_Wih_ + S_C_);
    int i = q0 / HUc, o = q0 - i * HUc;
    size_t q = (size_t)o * INc + i;
#pragma unroll
    for (int b = 0; b < 32; ++b) DT[(size_t)b * S_D_ + q] = (bf16)acc[b];
  }
}

// ---- K3: gx1[t,b,g] = inputs[b] @ Wih[b]  (per-batch WMMA GEMM) ---------
__global__ void gx1_kernel(const bf16* __restrict__ xb,    // (B,T,IN) bf16
                           const bf16* __restrict__ wihT,  // (B,3H,IN) bf16
                           float* __restrict__ gx) {       // (T,B,3H) f32
  int wave = threadIdx.x >> 5;                 // 8 waves / block
  int n0 = (blockIdx.x * 8 + wave) * 16;       // 96 n-tiles
  int t0 = blockIdx.y * 16;                    // 7 t-tiles
  int b  = blockIdx.z;                         // 32
  v8f acc = {};
#pragma unroll
  for (int k0 = 0; k0 < INc; k0 += 32) {
    v16bf a  = load_A(xb + ((size_t)b * Tc + t0) * INc + k0, INc, Tc - t0);
    v16bf bm = load_BT(wihT + (size_t)b * S_Wih_ + (size_t)n0 * INc + k0, INc);
    acc = wmma_bf16(a, bm, acc);
  }
  int l = threadIdx.x & 31;
  int n = l & 15, mh = (l >> 4) * 8;
#pragma unroll
  for (int r = 0; r < 8; ++r) {
    int t = t0 + mh + r;
    if (t < Tc) gx[((size_t)t * Bc + b) * G3 + n0 + n] = acc[r];
  }
}

// ---- K4: gx2[t,b,g] = inputs @ Wih2 + bih2  (shared-weight WMMA GEMM) ---
__global__ void gx2_kernel(const bf16* __restrict__ xb,     // (B*T, IN) bf16
                           const bf16* __restrict__ wih2T,  // (3H,IN) bf16
                           const float* __restrict__ bih2,
                           float* __restrict__ gx2) {       // (T,B,3H) f32
  int wave = threadIdx.x >> 5;                 // 8 waves / block
  int n0 = (blockIdx.x * 8 + wave) * 16;       // 96 n-tiles
  int r0 = blockIdx.y * 16;                    // 200 row tiles (b*T + t)
  v8f acc = {};
#pragma unroll
  for (int k0 = 0; k0 < INc; k0 += 32) {
    v16bf a  = load_A(xb + (size_t)r0 * INc + k0, INc, Bc * Tc - r0);
    v16bf bm = load_BT(wih2T + (size_t)n0 * INc + k0, INc);
    acc = wmma_bf16(a, bm, acc);
  }
  int l = threadIdx.x & 31;
  int n = l & 15, mh = (l >> 4) * 8;
  float bias = bih2[n0 + n];
#pragma unroll
  for (int r = 0; r < 8; ++r) {
    int g = r0 + mh + r;
    int b = g / Tc, t = g - b * Tc;
    gx2[((size_t)t * Bc + b) * G3 + n0 + n] = acc[r] + bias;
  }
}

// ---- K5: init recurrent states ------------------------------------------
__global__ void init_state_kernel(const float* __restrict__ state,
                                  float* __restrict__ h1f, bf16* __restrict__ h1b,
                                  float* __restrict__ h2f, bf16* __restrict__ h2b) {
  int i = blockIdx.x * blockDim.x + threadIdx.x;
  if (i < Bc * Hc) {
    float v = state[i];
    h1f[i] = v;        h1b[i] = (bf16)v;
    h2f[i] = 0.0f;     h2b[i] = (bf16)0.0f;
  }
}

// ---- K6: GRU1 one timestep (per-batch Whh^T, GEMV-as-WMMA, fused gates) -
__global__ void gru1_step_kernel(const bf16* __restrict__ whhT,  // (B,3H,H)
                                 const float* __restrict__ bhp,  // (B,3H)
                                 const float* __restrict__ gx,   // (T,B,3H)
                                 const float* __restrict__ hin,
                                 const bf16*  __restrict__ hinb,
                                 float* __restrict__ hout,
                                 bf16*  __restrict__ houtb,
                                 bf16*  __restrict__ hsb,        // (B,T,H)
                                 int t) {
  int b = blockIdx.y;                        // 32
  int wave = threadIdx.x >> 5;               // 4 waves / block
  int j0 = (blockIdx.x * 4 + wave) * 16;     // 32 j-tiles total
  __shared__ bf16 hsm[Hc];
  for (int i = threadIdx.x; i < Hc; i += blockDim.x) hsm[i] = hinb[b * Hc + i];
  __syncthreads();
  v8f ar = {}, az = {}, an = {};
  const bf16* wb = whhT + (size_t)b * S_Whh_;
  int lane = threadIdx.x & 31;
  int nb = lane & 15, khb = (lane >> 4) * 16;
  const bf16* rowr = wb + (size_t)(j0 + nb) * Hc + khb;
  const bf16* rowz = rowr + (size_t)Hc * Hc;
  const bf16* rown = rowz + (size_t)Hc * Hc;
#pragma unroll 4
  for (int k0 = 0; k0 < Hc; k0 += 32) {
    v16bf a = load_A_bcast(hsm + k0);
    v16bf br = load_BT(wb + (size_t)j0 * Hc + k0, Hc);
    v16bf bz = load_BT(wb + (size_t)(Hc + j0) * Hc + k0, Hc);
    v16bf bn = load_BT(wb + (size_t)(2 * Hc + j0) * Hc + k0, Hc);
    if (k0 + 32 < Hc) {                      // stream-prefetch next K slab
      __builtin_prefetch(rowr + k0 + 32, 0, 1);
      __builtin_prefetch(rowz + k0 + 32, 0, 1);
      __builtin_prefetch(rown + k0 + 32, 0, 1);
    }
    ar = wmma_bf16(a, br, ar);
    az = wmma_bf16(a, bz, az);
    an = wmma_bf16(a, bn, an);
  }
  int l = threadIdx.x & 31;
  if (l < 16) {                              // rows are all identical; use row 0
    int j = j0 + l;
    size_t gxo = ((size_t)t * Bc + b) * G3;
    const float* bhb = bhp + (size_t)b * G3;
    float r   = sigmoidf_(gx[gxo + j]          + ar[0] + bhb[j]);
    float z   = sigmoidf_(gx[gxo + Hc + j]     + az[0] + bhb[Hc + j]);
    float eta = tanhf(gx[gxo + 2 * Hc + j] + r * an[0] + bhb[2 * Hc + j]);
    float h  = hin[b * Hc + j];
    float hn = z * h + (1.0f - z) * eta;
    hout[b * Hc + j]  = hn;
    houtb[b * Hc + j] = (bf16)hn;
    hsb[((size_t)b * Tc + t) * Hc + j] = (bf16)hn;
  }
}

// ---- K7: GRU2 one timestep (shared Whh2^T, true M=32 WMMA GEMM) ---------
__global__ void gru2_step_kernel(const bf16* __restrict__ whh2T, // (3H,H)
                                 const float* __restrict__ bhh2, // (3H)
                                 const float* __restrict__ gx2,  // (T,B,3H)
                                 const float* __restrict__ hin,
                                 const bf16*  __restrict__ hinb,
                                 float* __restrict__ hout,
                                 bf16*  __restrict__ houtb,
                                 bf16*  __restrict__ hs2b,       // (B,T,H)
                                 int t) {
  int wave = threadIdx.x >> 5;               // 4 waves / block, 16 blocks
  int idx = blockIdx.x * 4 + wave;           // 0..63
  int mt = idx & 1;                          // batch tile (rows 0-15 / 16-31)
  int j0 = (idx >> 1) * 16;                  // 32 j-tiles
  v8f ar = {}, az = {}, an = {};
  const bf16* abase = hinb + (size_t)mt * 16 * Hc;
#pragma unroll 4
  for (int k0 = 0; k0 < Hc; k0 += 32) {
    v16bf a  = load_A(abase + k0, Hc, 16);
    v16bf br = load_BT(whh2T + (size_t)j0 * Hc + k0, Hc);
    v16bf bz = load_BT(whh2T + (size_t)(Hc + j0) * Hc + k0, Hc);
    v16bf bn = load_BT(whh2T + (size_t)(2 * Hc + j0) * Hc + k0, Hc);
    ar = wmma_bf16(a, br, ar);
    az = wmma_bf16(a, bz, az);
    an = wmma_bf16(a, bn, an);
  }
  int l = threadIdx.x & 31;
  int n = l & 15, mh = (l >> 4) * 8;
  int j = j0 + n;
  float br_ = bhh2[j], bz_ = bhh2[Hc + j], bn_ = bhh2[2 * Hc + j];
#pragma unroll
  for (int r = 0; r < 8; ++r) {
    int b = mt * 16 + mh + r;
    size_t gxo = ((size_t)t * Bc + b) * G3;
    float rr = sigmoidf_(gx2[gxo + j]      + ar[r] + br_);
    float zz = sigmoidf_(gx2[gxo + Hc + j] + az[r] + bz_);
    float nn = tanhf(gx2[gxo + 2 * Hc + j] + rr * (an[r] + bn_));
    float h  = hin[b * Hc + j];
    float hn = (1.0f - zz) * nn + zz * h;
    hout[b * Hc + j]  = hn;
    houtb[b * Hc + j] = (bf16)hn;
    hs2b[((size_t)b * Tc + t) * Hc + j] = (bf16)hn;
  }
}

// ---- K8: output projections; mode 0 -> yhats2, mode 1 -> yhats1 ---------
__global__ void yhat_kernel(const bf16* __restrict__ hsb,   // (B,T,H)
                            const bf16* __restrict__ hs2b,  // (B,T,H)
                            const bf16* __restrict__ xb,    // (B,T,IN)
                            const bf16* __restrict__ CT,    // (B,64,H)
                            const bf16* __restrict__ DT,    // (B,64,IN)
                            float* __restrict__ out) {      // (2B,T,64)
  int wave = threadIdx.x >> 5;               // 4 waves / block = 4 n-tiles
  int n0 = wave * 16;
  int t0 = blockIdx.y * 16;                  // 7 tiles
  int b = blockIdx.z & 31;
  int mode = blockIdx.z >> 5;                // uniform per block
  const bf16* A1 = (mode ? hsb : hs2b) + (size_t)b * Tc * Hc;
  const bf16* Cw = CT + (size_t)(mode ? b : Bc - 1) * S_C_;
  const bf16* Dw = DT + (size_t)(mode ? b : Bc - 1) * S_D_;
  v8f acc = {};
#pragma unroll 4
  for (int k0 = 0; k0 < Hc; k0 += 32) {
    v16bf a  = load_A(A1 + (size_t)t0 * Hc + k0, Hc, Tc - t0);
    v16bf bm = load_BT(Cw + (size_t)n0 * Hc + k0, Hc);
    acc = wmma_bf16(a, bm, acc);
  }
  const bf16* X = xb + (size_t)b * Tc * INc;
#pragma unroll
  for (int k0 = 0; k0 < INc; k0 += 32) {
    v16bf a  = load_A(X + (size_t)t0 * INc + k0, INc, Tc - t0);
    v16bf bm = load_BT(Dw + (size_t)n0 * INc + k0, INc);
    acc = wmma_bf16(a, bm, acc);
  }
  int l = threadIdx.x & 31;
  int n = l & 15, mh = (l >> 4) * 8;
  int row = (mode ? Bc + b : b);
#pragma unroll
  for (int r = 0; r < 8; ++r) {
    int t = t0 + mh + r;
    if (t < Tc) out[((size_t)row * Tc + t) * HUc + n0 + n] = acc[r];
  }
}

// ---- host-side launch ----------------------------------------------------
extern "C" void kernel_launch(void* const* d_in, const int* in_sizes, int n_in,
                              void* d_out, int out_size, void* d_ws, size_t ws_size,
                              hipStream_t stream) {
  const float* inputs = (const float*)d_in[0];
  const float* mu     = (const float*)d_in[1];
  const float* sd     = (const float*)d_in[2];
  const float* eps    = (const float*)d_in[3];
  const float* state  = (const float*)d_in[4];
  const float* psi_W1 = (const float*)d_in[5];
  const float* psi_b1 = (const float*)d_in[6];
  const float* psi_W2 = (const float*)d_in[7];
  const float* psi_b2 = (const float*)d_in[8];
  const float* psi_W3 = (const float*)d_in[9];
  const float* psi_b3 = (const float*)d_in[10];
  const float* Wih2   = (const float*)d_in[11];
  const float* Whh2   = (const float*)d_in[12];
  const float* bih2   = (const float*)d_in[13];
  const float* bhh2   = (const float*)d_in[14];
  float* out = (float*)d_out;

  char* w = (char*)d_ws;
  auto carve = [&](size_t bytes) -> void* {
    void* p = (void*)w;
    w += (bytes + 255) & ~(size_t)255;
    return p;
  };
  bf16*  xb     = (bf16*) carve((size_t)Bc * Tc * INc * 2);
  bf16*  wih2T  = (bf16*) carve((size_t)INc * G3 * 2);
  bf16*  whh2T  = (bf16*) carve((size_t)Hc * G3 * 2);
  float* h2mlp  = (float*)carve((size_t)Bc * 32 * 4);
  bf16*  whhT   = (bf16*) carve((size_t)Bc * S_Whh_ * 2);
  float* bhp    = (float*)carve((size_t)Bc * G3 * 4);
  bf16*  wihT   = (bf16*) carve((size_t)Bc * S_Wih_ * 2);
  bf16*  CT     = (bf16*) carve((size_t)Bc * S_C_ * 2);
  bf16*  DT     = (bf16*) carve((size_t)Bc * S_D_ * 2);
  float* gx1    = (float*)carve((size_t)Tc * Bc * G3 * 4);
  float* gx2    = (float*)carve((size_t)Tc * Bc * G3 * 4);
  float* h1f[2] = {(float*)carve((size_t)Bc * Hc * 4), (float*)carve((size_t)Bc * Hc * 4)};
  bf16*  h1b[2] = {(bf16*) carve((size_t)Bc * Hc * 2), (bf16*) carve((size_t)Bc * Hc * 2)};
  float* h2f[2] = {(float*)carve((size_t)Bc * Hc * 4), (float*)carve((size_t)Bc * Hc * 4)};
  bf16*  h2b[2] = {(bf16*) carve((size_t)Bc * Hc * 2), (bf16*) carve((size_t)Bc * Hc * 2)};
  bf16*  hsb    = (bf16*) carve((size_t)Bc * Tc * Hc * 2);
  bf16*  hs2b   = (bf16*) carve((size_t)Bc * Tc * Hc * 2);
  (void)ws_size; (void)n_in; (void)in_sizes; (void)out_size;

  // bf16 conversions; recurrent/projection weights stored transposed for
  // contiguous WMMA B-operand (b128) loads
  cvt_bf16_kernel<<<(Bc * Tc * INc + 255) / 256, 256, 0, stream>>>(inputs, xb, Bc * Tc * INc);
  cvt_T_bf16_kernel<<<(INc * G3 + 255) / 256, 256, 0, stream>>>(Wih2, wih2T, INc, G3);
  cvt_T_bf16_kernel<<<(Hc * G3 + 255) / 256, 256, 0, stream>>>(Whh2, whh2T, Hc, G3);

  // hypernetwork
  hyper_kernel<<<Bc, 512, 0, stream>>>(mu, sd, eps, psi_W1, psi_b1, psi_W2, psi_b2, h2mlp);
  psi_kernel<<<(Pp + 255) / 256, 256, 0, stream>>>(h2mlp, psi_W3, psi_b3,
                                                   whhT, bhp, wihT, CT, DT);

  // input projections
  gx1_kernel<<<dim3(12, 7, Bc), 256, 0, stream>>>(xb, wihT, gx1);
  gx2_kernel<<<dim3(12, (Bc * Tc) / 16), 256, 0, stream>>>(xb, wih2T, bih2, gx2);

  // recurrent scans
  init_state_kernel<<<(Bc * Hc + 255) / 256, 256, 0, stream>>>(state, h1f[0], h1b[0],
                                                               h2f[0], h2b[0]);
  for (int t = 0; t < Tc; ++t) {
    int cur = t & 1, nxt = cur ^ 1;
    gru1_step_kernel<<<dim3(8, Bc), 128, 0, stream>>>(whhT, bhp, gx1,
                                                      h1f[cur], h1b[cur],
                                                      h1f[nxt], h1b[nxt], hsb, t);
    gru2_step_kernel<<<16, 128, 0, stream>>>(whh2T, bhh2, gx2,
                                             h2f[cur], h2b[cur],
                                             h2f[nxt], h2b[nxt], hs2b, t);
  }

  // output projections (mode 0: yhats2 with C[-1],D[-1]; mode 1: yhats1)
  yhat_kernel<<<dim3(1, 7, 2 * Bc), 128, 0, stream>>>(hsb, hs2b, xb, CT, DT, out);
}
